// GPT2Block_90701119357003
// MI455X (gfx1250) — matrix operand order repaired and checked by reading, not compiled, not run on hardware
//
#include <hip/hip_runtime.h>
#include <math.h>

// ---------------------------------------------------------------------------
// GPT-2 block for MI455X (gfx1250): bf16 WMMA GEMMs + flash attention.
// ---------------------------------------------------------------------------

typedef __bf16 bf16;
typedef __attribute__((ext_vector_type(16))) __bf16 v16bf;
typedef __attribute__((ext_vector_type(8)))  float  v8f;
typedef __attribute__((ext_vector_type(4)))  unsigned int uint4v;

#define D_MODEL  1024
#define INNER    4096
#define S_LEN    2048
#define BATCH    2
#define N_HEADS  16
#define HEAD_DIM 64
#define TOKENS   (BATCH * S_LEN)

__device__ __forceinline__ bf16 us2bf(unsigned short u) {
    union { unsigned short s; bf16 b; } x; x.s = u; return x.b;
}

__device__ __forceinline__ v8f wmma_bf16(v16bf a, v16bf b, v8f c) {
    // 8 args: (neg_a, A, neg_b, B, c_mod, C, reuse_a, reuse_b)
    return __builtin_amdgcn_wmma_f32_16x16x32_bf16(false, a, false, b,
                                                   (short)0, c, false, false);
}

// Build a 16xbf16 fragment from two 16-byte-aligned 16-byte runs.
// (A 16x32 frag: elems 0..15 = K koff..koff+15, koff = 16*(lane>>4);
//  B 32x16 frag: elems 0..15 = K kbase..kbase+15 -> p1 = p0 + 8.)
__device__ __forceinline__ v16bf load_frag(const bf16* p0, const bf16* p1) {
    union { unsigned int u[8]; v16bf v; } f;
    *(uint4v*)&f.u[0] = *(const uint4v*)p0;
    *(uint4v*)&f.u[4] = *(const uint4v*)p1;
    return f.v;
}

// ---------------------------------------------------------------------------
// fp32 -> bf16 weight conversion (grid-stride)
// ---------------------------------------------------------------------------
__global__ void cvt_f32_bf16(const float* __restrict__ in, bf16* __restrict__ out, int n) {
    int i = blockIdx.x * blockDim.x + threadIdx.x;
    int stride = gridDim.x * blockDim.x;
    for (; i < n; i += stride) out[i] = (bf16)in[i];
}

// ---------------------------------------------------------------------------
// LayerNorm over last dim (D_MODEL=1024), one block (256 thr) per row,
// writes bf16 activations.
// ---------------------------------------------------------------------------
__global__ __launch_bounds__(256)
void ln_to_bf16(const float* __restrict__ x, const float* __restrict__ g,
                const float* __restrict__ be, bf16* __restrict__ out) {
    const int row = blockIdx.x;
    const int tid = threadIdx.x;
    const float* xr = x + (size_t)row * D_MODEL;

    float v[4];
    float s = 0.f;
#pragma unroll
    for (int i = 0; i < 4; ++i) { v[i] = xr[tid + 256 * i]; s += v[i]; }

    __shared__ float red[8];
#pragma unroll
    for (int off = 16; off > 0; off >>= 1) s += __shfl_xor(s, off);
    if ((tid & 31) == 0) red[tid >> 5] = s;
    __syncthreads();
    float tot = 0.f;
#pragma unroll
    for (int i = 0; i < 8; ++i) tot += red[i];
    const float mean = tot * (1.0f / (float)D_MODEL);
    __syncthreads();

    float s2 = 0.f;
#pragma unroll
    for (int i = 0; i < 4; ++i) { float d = v[i] - mean; s2 += d * d; }
#pragma unroll
    for (int off = 16; off > 0; off >>= 1) s2 += __shfl_xor(s2, off);
    if ((tid & 31) == 0) red[tid >> 5] = s2;
    __syncthreads();
    tot = 0.f;
#pragma unroll
    for (int i = 0; i < 8; ++i) tot += red[i];
    const float rstd = rsqrtf(tot * (1.0f / (float)D_MODEL) + 1e-5f);

#pragma unroll
    for (int i = 0; i < 4; ++i) {
        int c = tid + 256 * i;
        out[(size_t)row * D_MODEL + c] = (bf16)((v[i] - mean) * rstd * g[c] + be[c]);
    }
}

// ---------------------------------------------------------------------------
// bf16 WMMA GEMM: out[M,N] = A[M,K] @ W[K,N] + bias (+resid) (+gelu)
// Block tile 128x128, BK=32, 8 waves of 32x64 each (2x4 WMMA tiles).
// LDS stride 40 halfs (80 B) -> every fragment read is a 16-B aligned b128.
// ---------------------------------------------------------------------------
#define LDA 40

template <int ACT, bool OBF>
__global__ __launch_bounds__(256)
void gemm_bf16_kernel(const bf16* __restrict__ A, const bf16* __restrict__ W,
                      const float* __restrict__ bias, const float* __restrict__ resid,
                      void* __restrict__ outp, int M, int N, int K) {
    __shared__ bf16 As[128 * LDA];    // A tile, row-major [128][32]
    __shared__ bf16 WsT[128 * LDA];   // W tile, transposed [n][k]

    const int tid  = threadIdx.x;
    const int lane = tid & 31;
    const int wave = tid >> 5;
    const int ml   = lane & 15;
    const int hi16 = lane >> 4;
    const int m0 = blockIdx.y * 128;
    const int n0 = blockIdx.x * 128;
    const int wm = (wave >> 1) * 32;   // 4 wave-rows
    const int wn = (wave & 1) * 64;    // 2 wave-cols

    const v8f vzero = {0.f, 0.f, 0.f, 0.f, 0.f, 0.f, 0.f, 0.f};
    v8f acc[2][4];
#pragma unroll
    for (int i = 0; i < 2; ++i)
#pragma unroll
        for (int j = 0; j < 4; ++j) acc[i][j] = vzero;

    for (int k0 = 0; k0 < K; k0 += 32) {
        __syncthreads();
        // A tile: 128x32 bf16 = 2048 u32, 8 per thread
#pragma unroll
        for (int i = 0; i < 8; ++i) {
            int t   = tid + i * 256;
            int row = t >> 4;
            int col = (t & 15) << 1;
            *(unsigned int*)(&As[row * LDA + col]) =
                *(const unsigned int*)(A + (size_t)(m0 + row) * K + k0 + col);
        }
        // W tile: 32x128 bf16 loaded row-major, stored transposed [n][k]
#pragma unroll
        for (int i = 0; i < 8; ++i) {
            int t = tid + i * 256;
            int k = t >> 6;
            int n = (t & 63) << 1;
            unsigned int u = *(const unsigned int*)(W + (size_t)(k0 + k) * N + n0 + n);
            WsT[(n    ) * LDA + k] = us2bf((unsigned short)(u & 0xffffu));
            WsT[(n + 1) * LDA + k] = us2bf((unsigned short)(u >> 16));
        }
        // prefetch next k-tile while this one is consumed
        if (k0 + 32 < K) {
            __builtin_prefetch(A + (size_t)(m0 + (tid >> 1)) * K + k0 + 32 + (tid & 1) * 16, 0, 0);
            __builtin_prefetch(W + (size_t)(k0 + 32 + (tid >> 3)) * N + n0 + (tid & 7) * 16, 0, 0);
        }
        __syncthreads();

        // A fragments (16x32): two contiguous 16-B runs per lane
        v16bf af[2], bfr[4];
        const int koff = hi16 * 16;
#pragma unroll
        for (int mi = 0; mi < 2; ++mi) {
            const bf16* base = &As[(wm + mi * 16 + ml) * LDA + koff];
            af[mi] = load_frag(base, base + 8);
        }
        // B fragments (32x16): one contiguous 32-B run per lane
        const int kbase = hi16 * 16;
#pragma unroll
        for (int ni = 0; ni < 4; ++ni) {
            const bf16* base = &WsT[(wn + ni * 16 + ml) * LDA + kbase];
            bfr[ni] = load_frag(base, base + 8);
        }
#pragma unroll
        for (int mi = 0; mi < 2; ++mi)
#pragma unroll
            for (int ni = 0; ni < 4; ++ni)
                acc[mi][ni] = wmma_bf16(af[mi], bfr[ni], acc[mi][ni]);
    }

    // Epilogue: C layout -> lane holds col (lane&15), rows j + 8*(lane>>4)
#pragma unroll
    for (int mi = 0; mi < 2; ++mi) {
#pragma unroll
        for (int ni = 0; ni < 4; ++ni) {
            int c = n0 + wn + ni * 16 + ml;
            float bv = bias[c];
#pragma unroll
            for (int j = 0; j < 8; ++j) {
                int r = m0 + wm + mi * 16 + j + 8 * hi16;
                float v = acc[mi][ni][j] + bv;
                if (resid) v += resid[(size_t)r * N + c];
                if (ACT == 1) v = 0.5f * v * (1.0f + erff(v * 0.70710678118654752f));
                if (OBF) ((bf16*)outp)[(size_t)r * N + c] = (bf16)v;
                else     ((float*)outp)[(size_t)r * N + c] = v;
            }
        }
    }
}

// ---------------------------------------------------------------------------
// Flash attention, one wave (32 thr) per (batch, head, 16-query tile).
// qkv is bf16 [TOKENS, 3*D_MODEL]; ctx is bf16 [TOKENS, D_MODEL].
// ---------------------------------------------------------------------------
__global__ __launch_bounds__(32)
void attn_kernel(const bf16* __restrict__ qkv, bf16* __restrict__ ctx) {
    const int qt   = blockIdx.x;
    const int head = blockIdx.y;
    const int b    = blockIdx.z;
    const int lane = threadIdx.x;
    const int ml   = lane & 15;
    const int hi16 = lane >> 4;
    const int q0   = qt * 16;
    const int RS   = 3 * D_MODEL;

    const bf16* Qb = qkv + (size_t)b * S_LEN * RS + head * HEAD_DIM;
    const bf16* Kb = Qb + D_MODEL;
    const bf16* Vb = Qb + 2 * D_MODEL;

    // Q fragments: two 16x32 tiles over d = 0..31, 32..63
    v16bf qf[2];
    {
        const bf16* qrow = Qb + (size_t)(q0 + ml) * RS;
        const int koff = hi16 * 16;
#pragma unroll
        for (int hlf = 0; hlf < 2; ++hlf) {
            const bf16* p = qrow + hlf * 32 + koff;
            qf[hlf] = load_frag(p, p + 8);
        }
    }

    const v8f vzero = {0.f, 0.f, 0.f, 0.f, 0.f, 0.f, 0.f, 0.f};
    v8f o[4] = {vzero, vzero, vzero, vzero};   // 16q x 64d accumulator
    float mrow[8], lrow[8];                    // per-row running max / sum
#pragma unroll
    for (int j = 0; j < 8; ++j) { mrow[j] = -1e30f; lrow[j] = 0.f; }

    __shared__ float Ps[16 * 32];              // P tile staging (transpose bounce)

    const int kvend = q0 + 16;
    for (int kv0 = 0; kv0 < kvend; kv0 += 32) {
        // --- S = Q @ K^T for two 16-kv sub-tiles ---
        v8f s[2] = {vzero, vzero};
#pragma unroll
        for (int sub = 0; sub < 2; ++sub) {
            const int kvb = kv0 + sub * 16;
            const bf16* krow = Kb + (size_t)(kvb + ml) * RS + hi16 * 16;
#pragma unroll
            for (int hlf = 0; hlf < 2; ++hlf) {
                // B frag: col n = kv row, K dim = d (contiguous in K rows)
                const bf16* p = krow + hlf * 32;
                v16bf kf = load_frag(p, p + 8);
                s[sub] = wmma_bf16(qf[hlf], kf, s[sub]);
            }
        }
        // --- scale + causal mask (reference uses -10000) ---
        float sv0[8], sv1[8];
#pragma unroll
        for (int j = 0; j < 8; ++j) {
            int r  = q0 + j + 8 * hi16;
            int c0 = kv0 + ml;
            int c1 = kv0 + 16 + ml;
            sv0[j] = (c0 <= r) ? s[0][j] * 0.125f : -10000.0f;
            sv1[j] = (c1 <= r) ? s[1][j] * 0.125f : -10000.0f;
        }
        // --- online softmax stats (row = across the 16 lanes of each half) ---
        float alpha[8];
#pragma unroll
        for (int j = 0; j < 8; ++j) {
            float tmx = fmaxf(sv0[j], sv1[j]);
#pragma unroll
            for (int off = 1; off < 16; off <<= 1) tmx = fmaxf(tmx, __shfl_xor(tmx, off));
            float mn = fmaxf(mrow[j], tmx);
            alpha[j] = __expf(mrow[j] - mn);
            mrow[j]  = mn;
        }
        __syncthreads();
        float tsum[8];
#pragma unroll
        for (int j = 0; j < 8; ++j) {
            float p0 = __expf(sv0[j] - mrow[j]);
            float p1 = __expf(sv1[j] - mrow[j]);
            Ps[(j + 8 * hi16) * 32 + ml]      = p0;
            Ps[(j + 8 * hi16) * 32 + 16 + ml] = p1;
            float ts = p0 + p1;
#pragma unroll
            for (int off = 1; off < 16; off <<= 1) ts += __shfl_xor(ts, off);
            tsum[j] = ts;
        }
#pragma unroll
        for (int j = 0; j < 8; ++j) lrow[j] = lrow[j] * alpha[j] + tsum[j];
#pragma unroll
        for (int dt = 0; dt < 4; ++dt)
#pragma unroll
            for (int j = 0; j < 8; ++j) o[dt][j] *= alpha[j];
        __syncthreads();

        // --- P fragment (A layout) from LDS ---
        v16bf pf;
        {
            const int koff = hi16 * 16;
            const float* prow = &Ps[ml * 32];
#pragma unroll
            for (int j = 0; j < 4; ++j) {
                pf[2 * j]     = (bf16)prow[koff + 2 * j];
                pf[2 * j + 1] = (bf16)prow[koff + 2 * j + 1];
            }
#pragma unroll
            for (int j = 0; j < 4; ++j) {
                pf[8 + 2 * j] = (bf16)prow[8 + koff + 2 * j];
                pf[9 + 2 * j] = (bf16)prow[8 + koff + 2 * j + 1];
            }
        }
        // --- O += P @ V ---
        const int kb = kv0 + hi16 * 16;
#pragma unroll
        for (int dt = 0; dt < 4; ++dt) {
            v16bf vf;  // B frag: col n = d, K dim = kv rows (strided -> 16-bit loads)
            const int d = dt * 16 + ml;
#pragma unroll
            for (int j = 0; j < 8; ++j) {
                vf[2 * j]     = Vb[(size_t)(kb + 2 * j)     * RS + d];
                vf[2 * j + 1] = Vb[(size_t)(kb + 2 * j + 1) * RS + d];
            }
            o[dt] = wmma_bf16(pf, vf, o[dt]);
        }
    }

    // --- normalize + store ---
#pragma unroll
    for (int j = 0; j < 8; ++j) lrow[j] = 1.0f / lrow[j];
#pragma unroll
    for (int dt = 0; dt < 4; ++dt) {
#pragma unroll
        for (int j = 0; j < 8; ++j) {
            int r = q0 + j + 8 * hi16;
            int d = dt * 16 + ml;
            ctx[((size_t)(b * S_LEN + r)) * D_MODEL + head * HEAD_DIM + d] =
                (bf16)(o[dt][j] * lrow[j]);
        }
    }
}

// ---------------------------------------------------------------------------
// Launch
// ---------------------------------------------------------------------------
extern "C" void kernel_launch(void* const* d_in, const int* in_sizes, int n_in,
                              void* d_out, int out_size, void* d_ws, size_t ws_size,
                              hipStream_t stream) {
    const float* h      = (const float*)d_in[0];
    const float* w_qkv  = (const float*)d_in[1];
    const float* b_qkv  = (const float*)d_in[2];
    const float* g1     = (const float*)d_in[3];
    const float* be1    = (const float*)d_in[4];
    const float* w_proj = (const float*)d_in[5];
    const float* b_proj = (const float*)d_in[6];
    const float* g2     = (const float*)d_in[7];
    const float* be2    = (const float*)d_in[8];
    const float* w_fc   = (const float*)d_in[9];
    const float* b_fc   = (const float*)d_in[10];
    const float* w_mlp  = (const float*)d_in[11];
    const float* b_mlp  = (const float*)d_in[12];
    float* outp = (float*)d_out;

    char* ws = (char*)d_ws;
    bf16*  wqkv_bf  = (bf16*)(ws);                  // 6,291,456 B
    bf16*  wproj_bf = (bf16*)(ws + 6291456);        // 2,097,152 B
    bf16*  wfc_bf   = (bf16*)(ws + 8388608);        // 8,388,608 B
    bf16*  wmlp_bf  = (bf16*)(ws + 16777216);       // 8,388,608 B
    bf16*  xbuf     = (bf16*)(ws + 25165824);       // 8,388,608 B  (ln1 out, then ln2 out)
    bf16*  big      = (bf16*)(ws + 33554432);       // 33,554,432 B (qkv, then fc+gelu out)
    bf16*  ctx      = (bf16*)(ws + 67108864);       // 8,388,608 B
    float* h2       = (float*)(ws + 75497472);      // 16,777,216 B  -> total ~92.3 MB

    // 1) weights -> bf16
    cvt_f32_bf16<<<4096, 256, 0, stream>>>(w_qkv,  wqkv_bf,  D_MODEL * 3 * D_MODEL);
    cvt_f32_bf16<<<4096, 256, 0, stream>>>(w_proj, wproj_bf, D_MODEL * D_MODEL);
    cvt_f32_bf16<<<4096, 256, 0, stream>>>(w_fc,   wfc_bf,   D_MODEL * INNER);
    cvt_f32_bf16<<<4096, 256, 0, stream>>>(w_mlp,  wmlp_bf,  INNER * D_MODEL);

    // 2) ln1
    ln_to_bf16<<<TOKENS, 256, 0, stream>>>(h, g1, be1, xbuf);

    // 3) qkv = ln1(h) @ w_qkv + b_qkv  (bf16 out)
    gemm_bf16_kernel<0, true><<<dim3(3 * D_MODEL / 128, TOKENS / 128), 256, 0, stream>>>(
        xbuf, wqkv_bf, b_qkv, nullptr, big, TOKENS, 3 * D_MODEL, D_MODEL);

    // 4) causal attention -> ctx (bf16)
    attn_kernel<<<dim3(S_LEN / 16, N_HEADS, BATCH), 32, 0, stream>>>(big, ctx);

    // 5) h2 = ctx @ w_proj + b_proj + h   (fp32 out)
    gemm_bf16_kernel<0, false><<<dim3(D_MODEL / 128, TOKENS / 128), 256, 0, stream>>>(
        ctx, wproj_bf, b_proj, h, h2, TOKENS, D_MODEL, D_MODEL);

    // 6) ln2
    ln_to_bf16<<<TOKENS, 256, 0, stream>>>(h2, g2, be2, xbuf);

    // 7) fc = gelu(ln2(h2) @ w_fc + b_fc)  (bf16 out)
    gemm_bf16_kernel<1, true><<<dim3(INNER / 128, TOKENS / 128), 256, 0, stream>>>(
        xbuf, wfc_bf, b_fc, nullptr, big, TOKENS, INNER, D_MODEL);

    // 8) out = fc @ w_mlp + b_mlp + h2  (fp32 out)
    gemm_bf16_kernel<0, false><<<dim3(D_MODEL / 128, TOKENS / 128), 256, 0, stream>>>(
        big, wmlp_bf, b_mlp, h2, outp, TOKENS, D_MODEL, INNER);
}
